// EntityCompositeAttentionRNNModel_71313636982844
// MI455X (gfx1250) — matrix operand order repaired and checked by reading, not compiled, not run on hardware
//
#include <hip/hip_runtime.h>
#include <hip/hip_bf16.h>
#include <math.h>

// Problem constants
#define T_SEQ  100
#define BATCH  64
#define NINP_  400
#define NHID_  1150
#define NTOKEN_ 33278
#define NGATE  4600          // 4*NHID
#define KP     1152          // padded K (36 * 32) for bf16 WMMA
#define KT     36            // K tiles of 32
#define NGP    4608          // padded gate count (72 * 64)
#define NTOKP  33280         // padded vocab (520 * 64)
#define TBROWS (T_SEQ * BATCH)  // 6400

typedef __attribute__((ext_vector_type(16))) __bf16 v16bf;
typedef __attribute__((ext_vector_type(8)))  float  v8f;

// ---- gfx1250 async global->LDS path (guarded; falls back to reg staging) ----
#ifndef USE_ASYNC_LDS
#if defined(__has_builtin)
#  if __has_builtin(__builtin_amdgcn_global_load_async_to_lds_b128) && \
      __has_builtin(__builtin_amdgcn_s_wait_asynccnt)
#    define USE_ASYNC_LDS 1
#  else
#    define USE_ASYNC_LDS 0
#  endif
#else
#  define USE_ASYNC_LDS 0
#endif
#endif

#if USE_ASYNC_LDS
// builtin signature (from clang diagnostic): param0 = int4 addrspace(1)*, then lds ptr, imm offset, imm cpol
typedef int v4i_ __attribute__((vector_size(16)));
typedef v4i_ __attribute__((address_space(1))) *gvp_t;
typedef v4i_ __attribute__((address_space(3))) *lvp_t;
__device__ __forceinline__ void async_copy16(const unsigned short* g, unsigned short* l) {
  __builtin_amdgcn_global_load_async_to_lds_b128((gvp_t)g, (lvp_t)l, 0, 0);
}
#endif

__device__ __forceinline__ unsigned short f2bf(float f) {
  unsigned u = __float_as_uint(f);
  unsigned r = u + 0x7FFFu + ((u >> 16) & 1u);   // round-to-nearest-even
  return (unsigned short)(r >> 16);
}

// ---------------------------------------------------------------------------
// Embedding gather: xa_bf[row, 0:400] = emb[input[row]], [400:800] = emb[input2]
// ---------------------------------------------------------------------------
__global__ void embed_to_bf(const int* __restrict__ in1, const int* __restrict__ in2,
                            const float* __restrict__ embW, unsigned short* __restrict__ xa) {
  int idx = blockIdx.x * blockDim.x + threadIdx.x;
  if (idx >= TBROWS * KP) return;
  int row = idx / KP, col = idx % KP;
  float v = 0.f;
  if (col < NINP_)            v = embW[(size_t)in1[row] * NINP_ + col];
  else if (col < 2 * NINP_)   v = embW[(size_t)in2[row] * NINP_ + (col - NINP_)];
  xa[idx] = f2bf(v);
}

// ---------------------------------------------------------------------------
// Generic fp32 -> bf16 convert with zero padding: dst[Md, Kd] from src[Ms, Ks]
// ---------------------------------------------------------------------------
__global__ void cvt_pad_bf(const float* __restrict__ src, unsigned short* __restrict__ dst,
                           int Ms, int Ks, int Md, int Kd) {
  int idx = blockIdx.x * blockDim.x + threadIdx.x;
  if (idx >= Md * Kd) return;
  int m = idx / Kd, k = idx % Kd;
  float v = (m < Ms && k < Ks) ? src[(size_t)m * Ks + k] : 0.f;
  dst[idx] = f2bf(v);
}

// ---------------------------------------------------------------------------
// Init LSTM state
// ---------------------------------------------------------------------------
__global__ void init_state(const float* __restrict__ h0, const float* __restrict__ c0,
                           float* __restrict__ cst, unsigned short* __restrict__ hbf) {
  int idx = blockIdx.x * blockDim.x + threadIdx.x;
  if (idx >= BATCH * KP) return;
  int b = idx / KP, j = idx % KP;
  hbf[idx] = (j < NHID_) ? f2bf(h0[(size_t)b * NHID_ + j]) : (unsigned short)0;
  if (j < NHID_) cst[(size_t)b * NHID_ + j] = c0[(size_t)b * NHID_ + j];
}

// ---------------------------------------------------------------------------
// bf16 WMMA GEMM:  C[M,N](f32) (+)= A[M,KP](bf16) * B[N,KP](bf16)^T
// 128 threads = 4 waves, 64x64 C tile. lda = ldb = KP (compile-time).
// Async double-buffered LDS pipeline when available.
// ---------------------------------------------------------------------------
#define LDSW 40   // padded LDS row stride (elements): 80B rows, 16B aligned

__global__ void __launch_bounds__(128)
gemm_bf16_wmma(const unsigned short* __restrict__ A,
               const unsigned short* __restrict__ B,
               float* __restrict__ C,
               int ldc, int Nstore, int Nbias,
               const float* __restrict__ bias1, const float* __restrict__ bias2,
               int accum) {
#if USE_ASYNC_LDS
  __shared__ __align__(16) unsigned short As[2][64 * LDSW];
  __shared__ __align__(16) unsigned short Bs[2][64 * LDSW];
#else
  __shared__ __align__(16) unsigned short As[1][64 * LDSW];
  __shared__ __align__(16) unsigned short Bs[1][64 * LDSW];
#endif

  const int tid  = threadIdx.x;
  const int lane = tid & 31;
  const int wave = tid >> 5;                 // 0..3
  const int m0   = blockIdx.y * 64;
  const int n0   = blockIdx.x * 64;
  const int l16  = lane & 15;
  const int kg   = lane >> 4;                // lane group 0/1

  // staging coordinates: thread handles rows r0 and r0+32, fixed 8-elem column kc
  const int r0 = tid >> 2;
  const int kc = (tid & 3) << 3;
  const unsigned short* aSrc0 = A + (size_t)(m0 + r0)      * KP + kc;
  const unsigned short* aSrc1 = A + (size_t)(m0 + r0 + 32) * KP + kc;
  const unsigned short* bSrc0 = B + (size_t)(n0 + r0)      * KP + kc;
  const unsigned short* bSrc1 = B + (size_t)(n0 + r0 + 32) * KP + kc;
  const int d0 = r0 * LDSW + kc;
  const int d1 = (r0 + 32) * LDSW + kc;

  v8f acc[4];
  if (accum) {
    for (int nb = 0; nb < 4; ++nb)
      for (int r = 0; r < 8; ++r) {
        int m = m0 + wave * 16 + kg * 8 + r;
        int n = n0 + nb * 16 + l16;
        acc[nb][r] = C[(size_t)m * ldc + n];
      }
  } else {
    for (int nb = 0; nb < 4; ++nb)
      for (int r = 0; r < 8; ++r) acc[nb][r] = 0.f;
  }

  const int am = wave * 16 + l16;
  union F { uint4 q[2]; v16bf v; };
  auto compute = [&](const unsigned short* as, const unsigned short* bs) {
    F af;
    af.q[0] = *reinterpret_cast<const uint4*>(&as[am * LDSW + kg * 16]);
    af.q[1] = *reinterpret_cast<const uint4*>(&as[am * LDSW + kg * 16 + 8]);
    F b0, b1, b2, b3;   // preload all B fragments, then 4 back-to-back WMMAs
    b0.q[0] = *reinterpret_cast<const uint4*>(&bs[( 0 + l16) * LDSW + kg * 16]);
    b0.q[1] = *reinterpret_cast<const uint4*>(&bs[( 0 + l16) * LDSW + kg * 16 + 8]);
    b1.q[0] = *reinterpret_cast<const uint4*>(&bs[(16 + l16) * LDSW + kg * 16]);
    b1.q[1] = *reinterpret_cast<const uint4*>(&bs[(16 + l16) * LDSW + kg * 16 + 8]);
    b2.q[0] = *reinterpret_cast<const uint4*>(&bs[(32 + l16) * LDSW + kg * 16]);
    b2.q[1] = *reinterpret_cast<const uint4*>(&bs[(32 + l16) * LDSW + kg * 16 + 8]);
    b3.q[0] = *reinterpret_cast<const uint4*>(&bs[(48 + l16) * LDSW + kg * 16]);
    b3.q[1] = *reinterpret_cast<const uint4*>(&bs[(48 + l16) * LDSW + kg * 16 + 8]);
    acc[0] = __builtin_amdgcn_wmma_f32_16x16x32_bf16(false, af.v, false, b0.v, (short)0, acc[0], false, false);
    acc[1] = __builtin_amdgcn_wmma_f32_16x16x32_bf16(false, af.v, false, b1.v, (short)0, acc[1], false, false);
    acc[2] = __builtin_amdgcn_wmma_f32_16x16x32_bf16(false, af.v, false, b2.v, (short)0, acc[2], false, false);
    acc[3] = __builtin_amdgcn_wmma_f32_16x16x32_bf16(false, af.v, false, b3.v, (short)0, acc[3], false, false);
  };

#if USE_ASYNC_LDS
  // two-stage software pipeline: issue k+1 while computing k
  {
    async_copy16(aSrc0, &As[0][d0]);
    async_copy16(aSrc1, &As[0][d1]);
    async_copy16(bSrc0, &Bs[0][d0]);
    async_copy16(bSrc1, &Bs[0][d1]);
  }
  for (int kt = 0; kt < KT; ++kt) {
    const int buf = kt & 1;
    if (kt + 1 < KT) {
      const size_t kb = (size_t)(kt + 1) * 32;
      async_copy16(aSrc0 + kb, &As[1 - buf][d0]);
      async_copy16(aSrc1 + kb, &As[1 - buf][d1]);
      async_copy16(bSrc0 + kb, &Bs[1 - buf][d0]);
      async_copy16(bSrc1 + kb, &Bs[1 - buf][d1]);
      __builtin_amdgcn_s_wait_asynccnt(4);   // stage kt fully landed in LDS
    } else {
      __builtin_amdgcn_s_wait_asynccnt(0);
    }
    __syncthreads();
    compute(As[buf], Bs[buf]);
    __syncthreads();                         // LDS buffer reuse fence
  }
#else
  for (int kt = 0; kt < KT; ++kt) {
    const size_t kb = (size_t)kt * 32;
    // issue all 4 global loads before any LDS store (overlap latency)
    uint4 a0 = *reinterpret_cast<const uint4*>(aSrc0 + kb);
    uint4 a1 = *reinterpret_cast<const uint4*>(aSrc1 + kb);
    uint4 b0 = *reinterpret_cast<const uint4*>(bSrc0 + kb);
    uint4 b1 = *reinterpret_cast<const uint4*>(bSrc1 + kb);
    if (kt + 1 < KT) {
      __builtin_prefetch(aSrc0 + kb + 32, 0, 0);
      __builtin_prefetch(bSrc0 + kb + 32, 0, 0);
    }
    __syncthreads();
    *reinterpret_cast<uint4*>(&As[0][d0]) = a0;
    *reinterpret_cast<uint4*>(&As[0][d1]) = a1;
    *reinterpret_cast<uint4*>(&Bs[0][d0]) = b0;
    *reinterpret_cast<uint4*>(&Bs[0][d1]) = b1;
    __syncthreads();
    compute(As[0], Bs[0]);
  }
#endif

  // epilogue: optional bias, masked store (handles unpadded output widths)
  for (int nb = 0; nb < 4; ++nb) {
    int n = n0 + nb * 16 + l16;
    float bv = 0.f;
    if (n < Nbias) {
      if (bias1) bv += bias1[n];
      if (bias2) bv += bias2[n];
    }
    if (n < Nstore) {
      for (int r = 0; r < 8; ++r) {
        int m = m0 + wave * 16 + kg * 8 + r;
        C[(size_t)m * ldc + n] = acc[nb][r] + bv;
      }
    }
  }
}

// ---------------------------------------------------------------------------
// LSTM cell pointwise (gate order i,f,g,o at offsets 0,1150,2300,3450)
// ---------------------------------------------------------------------------
__global__ void lstm_cell(const float* __restrict__ gates, float* __restrict__ cst,
                          float* __restrict__ hs_t, unsigned short* __restrict__ hbf) {
  int idx = blockIdx.x * blockDim.x + threadIdx.x;
  if (idx >= BATCH * NHID_) return;
  int b = idx / NHID_, j = idx % NHID_;
  const float* g = gates + (size_t)b * NGP;
  float gi = g[j], gf = g[NHID_ + j], gg = g[2 * NHID_ + j], go = g[3 * NHID_ + j];
  float i = 1.f / (1.f + __expf(-gi));
  float f = 1.f / (1.f + __expf(-gf));
  float gt = tanhf(gg);
  float o = 1.f / (1.f + __expf(-go));
  float c = f * cst[idx] + i * gt;
  cst[idx] = c;
  float h = o * tanhf(c);
  hs_t[idx] = h;
  hbf[(size_t)b * KP + j] = f2bf(h);
  if (j < KP - NHID_) hbf[(size_t)b * KP + NHID_ + j] = 0;
}

// ---------------------------------------------------------------------------
// Causal attention over last layer's hs [T,B,H]; one block per (t,b)
// ---------------------------------------------------------------------------
__global__ void __launch_bounds__(128)
attention_kernel(const float* __restrict__ hs, unsigned short* __restrict__ attbf) {
  const int t = blockIdx.x, b = blockIdx.y;
  const int tid = threadIdx.x, lane = tid & 31, wave = tid >> 5;
  __shared__ float sc[T_SEQ];
  __shared__ float red[4];
  const float* rowt = hs + ((size_t)t * BATCH + b) * NHID_;

  for (int s = 0; s <= t; ++s) {
    const float* rows = hs + ((size_t)s * BATCH + b) * NHID_;
    float p = 0.f;
    for (int j = tid; j < NHID_; j += 128) p += rowt[j] * rows[j];
    for (int off = 16; off > 0; off >>= 1) p += __shfl_down(p, off, 32);
    if (lane == 0) red[wave] = p;
    __syncthreads();
    if (tid == 0) sc[s] = red[0] + red[1] + red[2] + red[3];
    __syncthreads();
  }
  if (tid == 0) {
    float mx = -3.0e38f;
    for (int s = 0; s <= t; ++s) mx = fmaxf(mx, sc[s]);
    float sum = 0.f;
    for (int s = 0; s <= t; ++s) { float e = __expf(sc[s] - mx); sc[s] = e; sum += e; }
    float inv = 1.f / sum;
    for (int s = 0; s <= t; ++s) sc[s] *= inv;
  }
  __syncthreads();
  unsigned short* orow = attbf + ((size_t)t * BATCH + b) * KP;
  for (int j = tid; j < NHID_; j += 128) {
    float a = 0.f;
    for (int s = 0; s <= t; ++s) a += sc[s] * hs[((size_t)s * BATCH + b) * NHID_ + j];
    orow[j] = f2bf(a);
  }
  if (tid < KP - NHID_) orow[NHID_ + tid] = 0;
}

// ---------------------------------------------------------------------------
// Host-side orchestration (graph-capture safe)
// ---------------------------------------------------------------------------
extern "C" void kernel_launch(void* const* d_in, const int* in_sizes, int n_in,
                              void* d_out, int out_size, void* d_ws, size_t ws_size,
                              hipStream_t stream) {
  (void)in_sizes; (void)n_in; (void)out_size; (void)ws_size;

  const int*   input  = (const int*)d_in[0];
  const int*   input2 = (const int*)d_in[1];
  const float* hinit[3] = { (const float*)d_in[2], (const float*)d_in[4], (const float*)d_in[6] };
  const float* cinit[3] = { (const float*)d_in[3], (const float*)d_in[5], (const float*)d_in[7] };
  const float* embW   = (const float*)d_in[8];
  const float* Wih[3] = { (const float*)d_in[9],  (const float*)d_in[13], (const float*)d_in[17] };
  const float* Whh[3] = { (const float*)d_in[10], (const float*)d_in[14], (const float*)d_in[18] };
  const float* bih[3] = { (const float*)d_in[11], (const float*)d_in[15], (const float*)d_in[19] };
  const float* bhh[3] = { (const float*)d_in[12], (const float*)d_in[16], (const float*)d_in[20] };
  const float* decW   = (const float*)d_in[21];
  const float* decb   = (const float*)d_in[22];
  float* out = (float*)d_out;

  char* p = (char*)d_ws;
  auto carve = [&](size_t bytes) { char* r = p; p += (bytes + 255) & ~(size_t)255; return r; };
  unsigned short* xa_bf  = (unsigned short*)carve((size_t)TBROWS * KP * 2);
  unsigned short* Wb     = (unsigned short*)carve((size_t)NGP * KP * 2);
  unsigned short* Whb    = (unsigned short*)carve((size_t)NGP * KP * 2);
  float*          Xproj  = (float*)carve((size_t)TBROWS * NGP * 4);
  float*          hsbuf  = (float*)carve((size_t)TBROWS * NHID_ * 4);
  float*          cstate = (float*)carve((size_t)BATCH * NHID_ * 4);
  unsigned short* hbf    = (unsigned short*)carve((size_t)BATCH * KP * 2);
  unsigned short* attbf  = (unsigned short*)carve((size_t)TBROWS * KP * 2);
  unsigned short* decbf  = (unsigned short*)carve((size_t)NTOKP * KP * 2);

  const int in_real[3] = { 2 * NINP_, NHID_, NHID_ };

  {  // embedding gather -> bf16
    int n = TBROWS * KP;
    embed_to_bf<<<(n + 255) / 256, 256, 0, stream>>>(input, input2, embW, xa_bf);
  }

  for (int l = 0; l < 3; ++l) {
    {  // convert + pad weights
      int n = NGP * KP;
      cvt_pad_bf<<<(n + 255) / 256, 256, 0, stream>>>(Wih[l], Wb,  NGATE, in_real[l], NGP, KP);
      cvt_pad_bf<<<(n + 255) / 256, 256, 0, stream>>>(Whh[l], Whb, NGATE, NHID_,      NGP, KP);
    }
    {  // Xproj = x @ Wih^T + (b_ih + b_hh), all timesteps
      dim3 grid(NGP / 64, TBROWS / 64);        // (72, 100)
      gemm_bf16_wmma<<<grid, 128, 0, stream>>>(xa_bf, Wb, Xproj, NGP,
                                               NGP, NGATE, bih[l], bhh[l], 0);
    }
    {  // state init
      int n = BATCH * KP;
      init_state<<<(n + 255) / 256, 256, 0, stream>>>(hinit[l], cinit[l], cstate, hbf);
    }
    for (int t = 0; t < T_SEQ; ++t) {          // sequential recurrence
      float* gates = Xproj + (size_t)t * BATCH * NGP;
      dim3 grid(NGP / 64, 1);
      gemm_bf16_wmma<<<grid, 128, 0, stream>>>(hbf, Whb, gates, NGP,
                                               NGP, 0, nullptr, nullptr, 1);
      int n = BATCH * NHID_;
      lstm_cell<<<(n + 255) / 256, 256, 0, stream>>>(
          gates, cstate, hsbuf + (size_t)t * BATCH * NHID_, hbf);
    }
    if (l < 2) {  // next layer input
      int n = TBROWS * KP;
      cvt_pad_bf<<<(n + 255) / 256, 256, 0, stream>>>(hsbuf, xa_bf, TBROWS, NHID_, TBROWS, KP);
    }
  }

  {  // causal attention
    dim3 grid(T_SEQ, BATCH);
    attention_kernel<<<grid, 128, 0, stream>>>(hsbuf, attbf);
  }
  {  // decoder weights -> bf16 padded
    int n = NTOKP * KP;
    cvt_pad_bf<<<(n + 255) / 256, 256, 0, stream>>>(decW, decbf, NTOKEN_, NHID_, NTOKP, KP);
  }
  {  // decoded = att @ dec_W^T + dec_b
    dim3 grid(NTOKP / 64, TBROWS / 64);        // (520, 100)
    gemm_bf16_wmma<<<grid, 128, 0, stream>>>(attbf, decbf, out, NTOKEN_,
                                             NTOKEN_, NTOKEN_, decb, nullptr, 0);
  }
}
